// DynamicAttention_11570641895993
// MI455X (gfx1250) — matrix operand order repaired and checked by reading, not compiled, not run on hardware
//
#include <hip/hip_runtime.h>

// ---------------------------------------------------------------------------
// Types for CDNA5 WMMA
// ---------------------------------------------------------------------------
typedef __attribute__((ext_vector_type(16))) __bf16 v16bf;
typedef __attribute__((ext_vector_type(8)))  __bf16 bf16x8;
typedef __attribute__((ext_vector_type(8)))  float  v8f;
typedef __attribute__((ext_vector_type(8)))  unsigned short ushort8;
typedef __attribute__((ext_vector_type(4)))  float  f32x4;

#define HH   16     // heads
#define DD   768    // embedding
#define HD   48     // head dim
#define QL   64
#define KL   64
#define NN   4
#define ECHUNKS 4
#define EPC  75     // 300 / 4

// float -> bf16 (round-to-nearest-even), returned in low 16 bits
__device__ __forceinline__ unsigned int bf16_rne(float f) {
  unsigned int u = __builtin_bit_cast(unsigned int, f);
  return (u + 0x7FFFu + ((u >> 16) & 1u)) >> 16;
}

// split x = hi + lo into bf16 bit patterns
__device__ __forceinline__ void bf16_split(float x, unsigned int& hb, unsigned int& lb) {
  hb = bf16_rne(x);
  const float fh = __builtin_bit_cast(float, hb << 16);
  lb = bf16_rne(x - fh);
}

// single-instruction relu: v_med3_f32(x, 0, +inf)
__device__ __forceinline__ float relu1(float x) {
  return __builtin_amdgcn_fmed3f(x, 0.0f,
                                 __builtin_bit_cast(float, 0x7F800000u));
}

// load a 16-element bf16 fragment as two 16-byte LDS chunks
__device__ __forceinline__ v16bf load_frag(const unsigned short* p,
                                           int off_lo, int off_hi) {
  ushort8 u0 = *reinterpret_cast<const ushort8*>(p + off_lo);
  ushort8 u1 = *reinterpret_cast<const ushort8*>(p + off_hi);
  bf16x8 b0 = __builtin_bit_cast(bf16x8, u0);
  bf16x8 b1 = __builtin_bit_cast(bf16x8, u1);
  return __builtin_shufflevector(b0, b1, 0,1,2,3,4,5,6,7,
                                         8,9,10,11,12,13,14,15);
}

// ---------------------------------------------------------------------------
// Kernel 1: tiled dense projections. One block = 64-row x 64-col output tile.
// Row tiles: Q rows 0..255 -> rt 0..3; K -> rt 4; V -> rt 5.  72 blocks.
// ---------------------------------------------------------------------------
__global__ __launch_bounds__(256)
void proj_kernel(const float* __restrict__ q, const float* __restrict__ k,
                 const float* __restrict__ v,
                 const float* __restrict__ Wq, const float* __restrict__ bq,
                 const float* __restrict__ Wk, const float* __restrict__ bk,
                 const float* __restrict__ Wv, const float* __restrict__ bv,
                 float* __restrict__ Qp, float* __restrict__ Kp,
                 float* __restrict__ Vp) {
  const int rt = blockIdx.x / 12;   // row tile
  const int ct = blockIdx.x % 12;   // col tile
  const float *src, *W, *b; float* dst;
  if (rt < 4)       { src = q + rt * 64 * DD; W = Wq; b = bq; dst = Qp + rt * 64 * DD; }
  else if (rt == 4) { src = k;                W = Wk; b = bk; dst = Kp; }
  else              { src = v;                W = Wv; b = bv; dst = Vp; }

  __shared__ float inT[64 * 65];    // padded: avoid 16-way bank conflicts
  __shared__ float WT[64 * 64];

  const int t  = threadIdx.x;
  const int tr = t >> 4, tc = t & 15;

  float acc[4][4];
#pragma unroll
  for (int j = 0; j < 4; ++j) {
    const float bj = b[ct * 64 + tc + 16 * j];
#pragma unroll
    for (int i = 0; i < 4; ++i) acc[i][j] = bj;
  }

  for (int dc = 0; dc < 12; ++dc) {
    __syncthreads();
#pragma unroll
    for (int i = 0; i < 16; ++i) {
      const int idx = t + 256 * i;
      const int r = idx >> 6, d = idx & 63;
      inT[r * 65 + d] = src[r * DD + dc * 64 + d];
      WT[idx]         = W[(dc * 64 + r) * DD + ct * 64 + d];  // WT[d_local][c]
    }
    __syncthreads();
#pragma unroll 8
    for (int dd = 0; dd < 64; ++dd) {
      float a[4], w[4];
#pragma unroll
      for (int i = 0; i < 4; ++i) a[i] = inT[(tr + 16 * i) * 65 + dd];
#pragma unroll
      for (int j = 0; j < 4; ++j) w[j] = WT[dd * 64 + tc + 16 * j];
#pragma unroll
      for (int i = 0; i < 4; ++i)
#pragma unroll
        for (int j = 0; j < 4; ++j) acc[i][j] = fmaf(a[i], w[j], acc[i][j]);
    }
  }
#pragma unroll
  for (int i = 0; i < 4; ++i)
#pragma unroll
    for (int j = 0; j < 4; ++j)
      dst[(tr + 16 * i) * DD + ct * 64 + tc + 16 * j] = acc[i][j];
}

// ---------------------------------------------------------------------------
// Kernel 2: main loop. One block per (e-chunk, n, h). 512 threads = 16 waves.
// A = Q tile (rows q, e-invariant, fragments in registers); B = broad_sum
// (cols k, rebuilt per e, double-buffered). D[q,k] -> relu (v_med3) ->
// column-sum S[k] via in-lane tree + ONE xor-16 shuffle; W in registers.
// ---------------------------------------------------------------------------
__global__ __launch_bounds__(512)
void attn_main(const float* __restrict__ Qp, const float* __restrict__ Kp,
               const float* __restrict__ gmemory, float* __restrict__ Wbuf) {
  const int blk   = blockIdx.x;
  const int chunk = blk >> 6;
  const int nh    = blk & 63;
  const int n = nh >> 4, h = nh & 15;
  const int t    = threadIdx.x;
  const int lane = t & 31;
  const int wave = t >> 5;
  const int lm   = lane & 15;   // tile column (k) / fragment row selector
  const int hf   = lane >> 4;   // half: +8 row offset, K-chunk selector
  const int qt   = wave >> 2;   // q tile
  const int kt   = wave & 3;    // k tile

  __shared__ __align__(16) unsigned short Bhi[2][64 * 64], Blo[2][64 * 64];
  __shared__ __align__(16) unsigned short Qhi[64 * 64],    Qlo[64 * 64];
  __shared__ float Spart[2][4 * 64];

  // ---- init: K row-chunk to registers, Q hi/lo to LDS ----
  // thread t owns 8 consecutive d: row = t>>3, d0 = (t&7)*8
  const int row = t >> 3;
  const int d0  = (t & 7) * 8;
  float Kreg[8];
  {
    ushort8 hv, lv;
#pragma unroll
    for (int i = 0; i < 8; ++i) {
      const int d = d0 + i;
      Kreg[i] = (d < HD) ? Kp[row * DD + h * HD + d] : 0.f;
      const float qv = (d < HD) ? Qp[(n * QL + row) * DD + h * HD + d] : 0.f;
      unsigned int hb, lb; bf16_split(qv, hb, lb);
      hv[i] = (unsigned short)hb; lv[i] = (unsigned short)lb;
    }
    *reinterpret_cast<ushort8*>(&Qhi[t * 8]) = hv;
    *reinterpret_cast<ushort8*>(&Qlo[t * 8]) = lv;
  }
  __syncthreads();

  // ---- A fragments (Q, e-invariant): 16-bit A 16x32 layout ----
  const int qrow = qt * 16 + lm;
  const int a0 = qrow * 64 + hf * 8;        // d-block 0
  const int a1 = qrow * 64 + 32 + hf * 8;   // d-block 1
  const v16bf qa_h0 = load_frag(Qhi, a0, a0 + 16);
  const v16bf qa_h1 = load_frag(Qhi, a1, a1 + 16);
  const v16bf qa_l0 = load_frag(Qlo, a0, a0 + 16);
  const v16bf qa_l1 = load_frag(Qlo, a1, a1 + 16);

  const int krow = kt * 16 + lm;
  const int b0 = krow * 64 + hf * 8;        // B frag bases
  const int b1 = krow * 64 + 32 + hf * 8;
  const int e0 = chunk * EPC;

  float Wacc[8];
#pragma unroll
  for (int j = 0; j < 8; ++j) Wacc[j] = 0.f;

  for (int ei = 0; ei < EPC; ++ei) {
    const int e = e0 + ei;
    const int p = ei & 1;
    const float* mrow = gmemory + (e * HH + h) * HD;
    if (ei + 1 < EPC && t < 2)   // hint next memory row -> global_prefetch_b8
      __builtin_prefetch(gmemory + ((e + 1) * HH + h) * HD + t * 32, 0, 1);

    // ---- build broad_sum = K + mem[e], bf16 hi/lo, one b128 store each ----
    {
      ushort8 hv, lv;
      if (d0 < HD) {
        const f32x4 m0 = *reinterpret_cast<const f32x4*>(mrow + d0);
        const f32x4 m1 = *reinterpret_cast<const f32x4*>(mrow + d0 + 4);
#pragma unroll
        for (int i = 0; i < 8; ++i) {
          const float a = Kreg[i] + ((i < 4) ? m0[i] : m1[i - 4]);
          unsigned int hb, lb; bf16_split(a, hb, lb);
          hv[i] = (unsigned short)hb; lv[i] = (unsigned short)lb;
        }
      } else {
#pragma unroll
        for (int i = 0; i < 8; ++i) { hv[i] = 0; lv[i] = 0; }
      }
      *reinterpret_cast<ushort8*>(&Bhi[p][t * 8]) = hv;
      *reinterpret_cast<ushort8*>(&Blo[p][t * 8]) = lv;
    }
    __syncthreads();   // build visible

    // ---- B fragments + 6x v_wmma_f32_16x16x32_bf16 (hi*hi + hi*lo + lo*hi)
    const v16bf bb_h0 = load_frag(&Bhi[p][0], b0, b0 + 16);
    const v16bf bb_h1 = load_frag(&Bhi[p][0], b1, b1 + 16);
    const v16bf bb_l0 = load_frag(&Blo[p][0], b0, b0 + 16);
    const v16bf bb_l1 = load_frag(&Blo[p][0], b1, b1 + 16);

    v8f acc = {0.f, 0.f, 0.f, 0.f, 0.f, 0.f, 0.f, 0.f};
    acc = __builtin_amdgcn_wmma_f32_16x16x32_bf16(false, qa_h0, false, bb_h0, (short)0, acc, false, false);
    acc = __builtin_amdgcn_wmma_f32_16x16x32_bf16(false, qa_h1, false, bb_h1, (short)0, acc, false, false);
    acc = __builtin_amdgcn_wmma_f32_16x16x32_bf16(false, qa_h0, false, bb_l0, (short)0, acc, false, false);
    acc = __builtin_amdgcn_wmma_f32_16x16x32_bf16(false, qa_h1, false, bb_l1, (short)0, acc, false, false);
    acc = __builtin_amdgcn_wmma_f32_16x16x32_bf16(false, qa_l0, false, bb_h0, (short)0, acc, false, false);
    acc = __builtin_amdgcn_wmma_f32_16x16x32_bf16(false, qa_l1, false, bb_h1, (short)0, acc, false, false);

    // ---- relu (v_med3_f32); column-sum over q: tree + one xor-16 shuffle --
    float m[8];
#pragma unroll
    for (int j = 0; j < 8; ++j) m[j] = relu1(acc[j]);
    float s01 = m[0] + m[1], s23 = m[2] + m[3];
    float s45 = m[4] + m[5], s67 = m[6] + m[7];
    float inl = (s01 + s23) + (s45 + s67);
    const float scol = inl + __shfl_xor(inl, 16);
    // both halves hold identical scol: store unguarded (same value, same
    // address from lanes l and l^16) -> no exec-mask diamond
    Spart[p][qt * 64 + kt * 16 + lm] = scol;
    __syncthreads();   // S partials visible

    // ---- W[q][k] += S[k] * M[q][k]  -> pure register accumulation ----
    const float s = Spart[p][kt * 16 + lm] + Spart[p][64 + kt * 16 + lm] +
                    Spart[p][128 + kt * 16 + lm] + Spart[p][192 + kt * 16 + lm];
#pragma unroll
    for (int j = 0; j < 8; ++j) Wacc[j] = fmaf(s, m[j], Wacc[j]);
  }

  // ---- write partial W, transposed layout W_T[q][k] (coalesced over lm) ----
  float* Wout = Wbuf + (chunk * 64 + nh) * 4096;
#pragma unroll
  for (int j = 0; j < 8; ++j)
    Wout[(qt * 16 + hf * 8 + j) * 64 + kt * 16 + lm] = Wacc[j];
}

// ---------------------------------------------------------------------------
// Kernel 3: out[n,q,h,:] = sum_k (sum_chunks W_T[q,k]) * V[k,h,:]
// ---------------------------------------------------------------------------
__global__ __launch_bounds__(256)
void attn_final(const float* __restrict__ Wbuf, const float* __restrict__ Vp,
                float* __restrict__ out) {
  const int nh = blockIdx.x;
  const int n = nh >> 4, h = nh & 15;
  __shared__ float Wsum[64 * 64];   // [q][k]
  __shared__ float Vsh[64 * HD];    // [k][d]
  const int t = threadIdx.x;

#pragma unroll
  for (int i = 0; i < 16; ++i) {
    const int idx = t + i * 256;
    Wsum[idx] = Wbuf[nh * 4096 + idx] + Wbuf[(64 + nh) * 4096 + idx] +
                Wbuf[(128 + nh) * 4096 + idx] + Wbuf[(192 + nh) * 4096 + idx];
  }
#pragma unroll
  for (int i = 0; i < 12; ++i) {
    const int idx = t + i * 256;
    const int kk = idx / HD, d = idx % HD;
    Vsh[idx] = Vp[kk * DD + h * HD + d];
  }
  __syncthreads();

#pragma unroll
  for (int i = 0; i < 12; ++i) {
    const int idx = t + i * 256;
    const int qq = idx / HD, d = idx % HD;
    float s = 0.f;
    for (int kk = 0; kk < KL; ++kk)
      s = fmaf(Wsum[qq * 64 + kk], Vsh[kk * HD + d], s);
    out[(n * QL + qq) * DD + h * HD + d] = s;
  }
}

// ---------------------------------------------------------------------------
extern "C" void kernel_launch(void* const* d_in, const int* in_sizes, int n_in,
                              void* d_out, int out_size, void* d_ws, size_t ws_size,
                              hipStream_t stream) {
  const float* q      = (const float*)d_in[0];
  const float* k      = (const float*)d_in[1];
  const float* v      = (const float*)d_in[2];
  const float* Wq     = (const float*)d_in[3];
  const float* bq     = (const float*)d_in[4];
  const float* Wk     = (const float*)d_in[5];
  const float* bk     = (const float*)d_in[6];
  const float* Wv     = (const float*)d_in[7];
  const float* bv     = (const float*)d_in[8];
  const float* memory = (const float*)d_in[9];

  float* ws   = (float*)d_ws;
  float* Qp   = ws;                       // 256*768 = 196608 floats
  float* Kp   = ws + 196608;              //  64*768 =  49152
  float* Vp   = ws + 245760;              //  64*768 =  49152
  float* Wbuf = ws + 294912;              // 4*64*4096 = 1048576 floats

  proj_kernel<<<dim3(72), dim3(256), 0, stream>>>(q, k, v, Wq, bq, Wk, bk,
                                                  Wv, bv, Qp, Kp, Vp);
  attn_main<<<dim3(ECHUNKS * NN * HH), dim3(512), 0, stream>>>(Qp, Kp, memory, Wbuf);
  attn_final<<<dim3(NN * HH), dim3(256), 0, stream>>>(Wbuf, Vp, (float*)d_out);
}
